// InfiniteMixturePrototype2_79517024518218
// MI455X (gfx1250) — matrix operand order repaired and checked by reading, not compiled, not run on hardware
//
#include <hip/hip_runtime.h>
#include <hip/hip_bf16.h>

// ---------------------------------------------------------------------------
// Problem constants (from reference): B=1, N=4096, M=4096, D=512 (rd=1024),
// C=256, L=64. Output: (loss, acc) -> 2 floats.
// ---------------------------------------------------------------------------
#define NN   4096
#define MM   4096
#define D2   1024
#define CC   256
#define LL   64

typedef __attribute__((ext_vector_type(16))) __bf16 v16bf;
typedef __attribute__((ext_vector_type(8)))  float  v8f;

__device__ __forceinline__ unsigned short f2bf(float f) {
    unsigned u = __float_as_uint(f);
    unsigned r = u + 0x7FFFu + ((u >> 16) & 1u);   // round-to-nearest-even
    return (unsigned short)(r >> 16);
}

// orderable-uint mapping for float (monotonic): larger float -> larger uint
__device__ __forceinline__ unsigned ford(float f) {
    unsigned u = __float_as_uint(f);
    return (u & 0x80000000u) ? ~u : (u | 0x80000000u);
}
__device__ __forceinline__ float funord(unsigned o) {
    return __uint_as_float((o & 0x80000000u) ? (o & 0x7FFFFFFFu) : ~o);
}

// ---------------------------------------------------------------------------
// Compile-time-strided bf16 WMMA GEMM:
//   Out[i,j] = scale_i * sum_k A[i*SAI + k*SAK] * B[k*SBK + j*SBJ]
// Block: 256 threads = 8 waves. Block tile 64(M) x 64(N); each wave owns a
// 16x32 output slab (one A fragment reused across two B fragments -> two
// v_wmma_f32_16x16x32_bf16 per K-step). K-step 32.
// Double-buffered LDS tiles: one barrier per K-step; staging of tile k+1
// overlaps ds_load+WMMA of tile k.
// Requires M % 64 == 0, N % 64 == 0, K % 32 == 0 (true for both GEMMs).
// ---------------------------------------------------------------------------
template<long SAI, long SAK, long SBK, long SBJ, long LDO, bool SCALE>
__global__ __launch_bounds__(256)
void wmma_gemm_bf16(const float* __restrict__ A,
                    const float* __restrict__ B,
                    float* __restrict__ Out,
                    const float* __restrict__ rowScale, int K)
{
    constexpr int BM = 64, BN = 64, KT = 32;
    __shared__ unsigned short As[2][BM][KT];   // [buf][row][k] (64B rows)
    __shared__ unsigned short Bs[2][BN][KT];   // [buf][col][k] (transposed)

    const int t    = threadIdx.x;
    const int lane = t & 31;
    const int w    = t >> 5;
    const int mr   = lane & 15;     // row/col within 16
    const int lg   = lane >> 4;     // half-wave group (K split)

    const int rowBase = blockIdx.y * BM;
    const int colBase = blockIdx.x * BN;
    const int wm  = (w & 3) << 4;   // wave tile row offset in block tile
    const int wn  = (w >> 2) << 5;  // wave slab col offset (0 or 32)

    // thread-contiguous staging mapping: thread t owns row sr = t>>2,
    // k-run sk0 = (t&3)*8 .. +7  -> vectorizable when stride==1
    const int sr  = t >> 2;
    const int sk0 = (t & 3) << 3;

    auto stage = [&](int buf, int k0) {
        const float* ap = A + (long)(rowBase + sr) * SAI + (long)(k0 + sk0) * SAK;
        const float* bp = B + (long)(colBase + sr) * SBJ + (long)(k0 + sk0) * SBK;
        #pragma unroll
        for (int i = 0; i < 8; ++i)
            As[buf][sr][sk0 + i] = f2bf(ap[i * SAK]);
        #pragma unroll
        for (int i = 0; i < 8; ++i)
            Bs[buf][sr][sk0 + i] = f2bf(bp[i * SBK]);
    };

    v8f acc0 = {0.f, 0.f, 0.f, 0.f, 0.f, 0.f, 0.f, 0.f};
    v8f acc1 = {0.f, 0.f, 0.f, 0.f, 0.f, 0.f, 0.f, 0.f};

    const int nk = K / KT;
    stage(0, 0);
    int cur = 0;

    for (int kt = 0; kt < nk; ++kt) {
        __syncthreads();   // staged tile `cur` visible; prev reads of `cur` done

        // ---- gather fragments per CDNA5 WMMA register layout (buffer cur) ----
        // A 16x32 bf16: lanes 0-15 M=mr, K {lg*8..+7} U {16+lg*8..+7}
        union { uint4 q[2]; v16bf v; } fa, fb0, fb1;
        const uint4* ar  = reinterpret_cast<const uint4*>(&As[cur][wm + mr][0]);
        fa.q[0] = ar[lg];
        fa.q[1] = ar[2 + lg];
        // B 32x16 bf16: lanes N=mr, K {lg*8..+7} U {16+lg*8..+7}
        const uint4* br0 = reinterpret_cast<const uint4*>(&Bs[cur][wn + mr][0]);
        const uint4* br1 = reinterpret_cast<const uint4*>(&Bs[cur][wn + 16 + mr][0]);
        fb0.q[0] = br0[lg]; fb0.q[1] = br0[2 + lg];
        fb1.q[0] = br1[lg]; fb1.q[1] = br1[2 + lg];

        // ---- overlap: stage next tile into the other buffer ----
        if (kt + 1 < nk) {
            stage(cur ^ 1, (kt + 1) * KT);
            if (kt + 2 < nk) {  // pull tile kt+2 toward the WGP
                __builtin_prefetch(A + (long)(rowBase + sr) * SAI + (long)((kt + 2) * KT + sk0) * SAK, 0, 1);
                __builtin_prefetch(B + (long)(colBase + sr) * SBJ + (long)((kt + 2) * KT + sk0) * SBK, 0, 1);
            }
        }

        acc0 = __builtin_amdgcn_wmma_f32_16x16x32_bf16(
                   false, fa.v, false, fb0.v, (short)0, acc0, false, false);
        acc1 = __builtin_amdgcn_wmma_f32_16x16x32_bf16(
                   false, fa.v, false, fb1.v, (short)0, acc1, false, false);
        cur ^= 1;
    }

    // ---- epilogue: C/D layout: VGPR v -> row lg*8+v, col mr ----
    float sc[8];
    if (SCALE) {
        const float4* sp = reinterpret_cast<const float4*>(rowScale + rowBase + wm + lg * 8);
        float4 s0 = sp[0], s1 = sp[1];
        sc[0] = s0.x; sc[1] = s0.y; sc[2] = s0.z; sc[3] = s0.w;
        sc[4] = s1.x; sc[5] = s1.y; sc[6] = s1.z; sc[7] = s1.w;
    }
    float* op = Out + (long)(rowBase + wm + lg * 8) * LDO + (colBase + wn + mr);
    #pragma unroll
    for (int v = 0; v < 8; ++v) {
        float v0 = acc0[v], v1 = acc1[v];
        if (SCALE) { v0 *= sc[v]; v1 *= sc[v]; }
        op[0]  = v0;
        op[16] = v1;
        op += LDO;
    }
}

// ---------------------------------------------------------------------------
// invPS[c] = 1 / sum_n probs[n,c]   (guard: sum==0 -> 1)
// ---------------------------------------------------------------------------
__global__ __launch_bounds__(256)
void col_sum_inv(const float* __restrict__ probs, float* __restrict__ invPS)
{
    __shared__ float red[256];
    int c = blockIdx.x;
    float s = 0.f;
    for (int n = threadIdx.x; n < NN; n += 256)
        s += probs[(long)n * CC + c];
    red[threadIdx.x] = s;
    __syncthreads();
    for (int o = 128; o > 0; o >>= 1) {
        if (threadIdx.x < o) red[threadIdx.x] += red[threadIdx.x + o];
        __syncthreads();
    }
    if (threadIdx.x == 0) {
        float ps = red[0];
        invPS[c] = (ps == 0.f) ? 1.f : (1.f / ps);
    }
}

// ---------------------------------------------------------------------------
// out[row] = sum_j x[row, j]^2   (cols = 1024 here)
// ---------------------------------------------------------------------------
__global__ __launch_bounds__(256)
void row_sumsq(const float* __restrict__ x, int cols, float* __restrict__ out)
{
    __shared__ float red[256];
    int row = blockIdx.x;
    float s = 0.f;
    for (int j = threadIdx.x; j < cols; j += 256) {
        float v = x[(long)row * cols + j];
        s += v * v;
    }
    red[threadIdx.x] = s;
    __syncthreads();
    for (int o = 128; o > 0; o >>= 1) {
        if (threadIdx.x < o) red[threadIdx.x] += red[threadIdx.x + o];
        __syncthreads();
    }
    if (threadIdx.x == 0) out[row] = red[0];
}

__global__ void zero_out(float* out) {
    if (threadIdx.x < 2) out[threadIdx.x] = 0.f;
}

// ---------------------------------------------------------------------------
// Per-row (m) loss epilogue. One block of 256 threads per row; thread c
// handles logit[m,c]. Argmax tie-breaks (first index) are reproduced with
// packed keys: (orderable(val) << 32) | (0xFFFFFFFF - c), reduced by
// atomicMax in LDS (ds_max_u64 path).
// ---------------------------------------------------------------------------
__global__ __launch_bounds__(256)
void loss_epilogue(const float* __restrict__ cross,
                   const float* __restrict__ d2,
                   const float* __restrict__ p2,
                   const float* __restrict__ radii,
                   const int*   __restrict__ y_val,
                   const int*   __restrict__ slab,
                   float* __restrict__ out)
{
    __shared__ unsigned long long slots[LL];   // best-in-class keys
    __shared__ unsigned long long gslot;       // global argmax key
    __shared__ float red[256];

    const int m = blockIdx.x;
    const int c = threadIdx.x;

    if (c < LL) slots[c] = 0ull;
    if (c == 0) gslot = 0ull;
    __syncthreads();

    float r     = radii[c];
    float dist  = d2[m] + p2[c] - 2.f * cross[(long)m * CC + c];
    float logit = -dist / (2.f * r) - 512.f * logf(r);   // 0.5*dim = 512

    unsigned long long key =
        ((unsigned long long)ford(logit) << 32) |
        (unsigned long long)(0xFFFFFFFFu - (unsigned)c);

    int lab = slab[c];
    atomicMax(&slots[lab], key);
    atomicMax(&gslot, key);
    __syncthreads();

    unsigned long long g = gslot;
    float maxV = funord((unsigned)(g >> 32));    // marked-set max == global max

    bool marked = ((0xFFFFFFFFu - (unsigned)(slots[lab] & 0xFFFFFFFFull)) == (unsigned)c);
    red[c] = marked ? expf(logit - maxV) : 0.f;
    __syncthreads();
    for (int o = 128; o > 0; o >>= 1) {
        if (c < o) red[c] += red[c + o];
        __syncthreads();
    }

    if (c == 0) {
        float lse = maxV + logf(red[0]);
        int lt = y_val[m];                             // in [0, L)
        float tgt = funord((unsigned)(slots[lt] >> 32));
        atomicAdd(out + 0, (lse - tgt) * (1.0f / (float)MM));

        unsigned gidx = 0xFFFFFFFFu - (unsigned)(g & 0xFFFFFFFFull);
        int ypred = slab[gidx];
        atomicAdd(out + 1, (ypred == y_val[m]) ? 1.0f : 0.0f);
    }
}

// ---------------------------------------------------------------------------
extern "C" void kernel_launch(void* const* d_in, const int* in_sizes, int n_in,
                              void* d_out, int out_size, void* d_ws, size_t ws_size,
                              hipStream_t stream)
{
    (void)in_sizes; (void)n_in; (void)out_size; (void)ws_size;

    const float* h      = (const float*)d_in[0];   // (N, 2, 512) -> (N, 1024)
    const float* probs  = (const float*)d_in[1];   // (N, C)
    const float* h_val  = (const float*)d_in[2];   // (M, 1024)
    const float* radii  = (const float*)d_in[3];   // (C)
    const int*   y_val  = (const int*)d_in[4];     // (M)
    const int*   slab   = (const int*)d_in[5];     // (C)
    float* out = (float*)d_out;

    // workspace layout (floats)
    float* ws     = (float*)d_ws;
    float* protos = ws;                                  // C x 1024
    float* cross  = protos + (long)CC * D2;              // M x C
    float* invPS  = cross + (long)MM * CC;               // C
    float* d2     = invPS + CC;                          // M
    float* p2     = d2 + MM;                             // C

    zero_out<<<1, 64, 0, stream>>>(out);

    // 1) inverse prototype mass
    col_sum_inv<<<CC, 256, 0, stream>>>(probs, invPS);

    // 2) protos[c, rd] = (1/ps[c]) * sum_n probs[n,c] * h[n, rd]
    //    A(i=c, k=n) = probs[n*C + c] : SAI=1,    SAK=C
    //    B(k=n, j=rd) = h[n*1024+rd]  : SBK=1024, SBJ=1
    {
        dim3 grid(D2 / 64, CC / 64);
        wmma_gemm_bf16<1L, (long)CC, (long)D2, 1L, (long)D2, true>
            <<<grid, 256, 0, stream>>>(probs, h, protos, invPS, NN);
    }

    // 3) squared norms
    row_sumsq<<<MM, 256, 0, stream>>>(h_val, D2, d2);
    row_sumsq<<<CC, 256, 0, stream>>>(protos, D2, p2);

    // 4) cross[m, c] = sum_k h_val[m,k] * protos[c,k]
    //    A(i=m, k) = h_val[m*1024+k] : SAI=1024, SAK=1
    //    B(k, j=c) = protos[c*1024+k]: SBK=1,    SBJ=1024
    {
        dim3 grid(CC / 64, MM / 64);
        wmma_gemm_bf16<(long)D2, 1L, 1L, (long)D2, (long)CC, false>
            <<<grid, 256, 0, stream>>>(h_val, protos, cross, nullptr, D2);
    }

    // 5) logits + OR-loss + accuracy
    loss_epilogue<<<MM, 256, 0, stream>>>(cross, d2, p2, radii, y_val, slab, out);
}
